// Wav2VecLoss_19756849562143
// MI455X (gfx1250) — matrix operand order repaired and checked by reading, not compiled, not run on hardware
//
#include <hip/hip_runtime.h>
#include <hip/hip_bf16.h>

// ---------------- types ----------------
typedef __bf16 v16bf __attribute__((ext_vector_type(16)));
typedef float  v8f   __attribute__((ext_vector_type(8)));
typedef unsigned int u32x4 __attribute__((ext_vector_type(4)));
typedef int i32x4 __attribute__((ext_vector_type(4)));
typedef int i32x8 __attribute__((ext_vector_type(8)));

union BfVec { v16bf v; uint4 q[2]; };

#define CDIM 512
#define LDIM 1024
#define NT   32          // t-columns per block
#define LDSB_STRIDE 1040 // 512 bf16 + 16B pad (TDM pad: interval=256 dw, amount=4 dw)

#if defined(__has_builtin)
#if __has_builtin(__builtin_amdgcn_tensor_load_to_lds) && \
    __has_builtin(__builtin_amdgcn_s_wait_tensorcnt)
#define HAVE_TDM 1
#endif
#endif

__device__ __forceinline__ unsigned short f2bf(float f) {
  unsigned u = __float_as_uint(f);
  u += 0x7FFFu + ((u >> 16) & 1u);        // round-to-nearest-even
  return (unsigned short)(u >> 16);
}

__device__ __forceinline__ float log_sigmoid(float x) {
  return fminf(x, 0.f) - log1pf(expf(-fabsf(x)));
}

__device__ __forceinline__ unsigned pcg(unsigned x) {
  x = x * 747796405u + 2891336453u;
  unsigned w = ((x >> ((x >> 28u) + 4u)) ^ x) * 277803737u;
  return (w >> 22u) ^ w;
}

// ---------------- prep kernels ----------------
__global__ void k_conv_w_bf16(const float* __restrict__ src,
                              unsigned short* __restrict__ dst, int n) {
  int i = blockIdx.x * blockDim.x + threadIdx.x;
  if (i < n) dst[i] = f2bf(src[i]);
}

// [B][C][L] f32 -> [B][L][C] bf16 (tiled transpose)
__global__ void k_transpose_c_bf16(const float* __restrict__ src,
                                   unsigned short* __restrict__ dst,
                                   int C, int L) {
  __shared__ float tile[32][33];
  const int b  = blockIdx.z;
  const int t0 = blockIdx.x * 32, c0 = blockIdx.y * 32;
  const int lx = threadIdx.x, ly = threadIdx.y;
  const float* s = src + (size_t)b * C * L;
  #pragma unroll
  for (int i = 0; i < 32; i += 8)
    tile[ly + i][lx] = s[(size_t)(c0 + ly + i) * L + t0 + lx];
  __syncthreads();
  unsigned short* d = dst + (size_t)b * L * C;
  #pragma unroll
  for (int i = 0; i < 32; i += 8)
    d[(size_t)(t0 + ly + i) * C + c0 + lx] = f2bf(tile[lx][ly + i]);
}

// [B][C][L] f32 -> [B][L][C] f32 (tiled transpose)
__global__ void k_transpose_z_f32(const float* __restrict__ src,
                                  float* __restrict__ dst,
                                  int C, int L) {
  __shared__ float tile[32][33];
  const int b  = blockIdx.z;
  const int t0 = blockIdx.x * 32, c0 = blockIdx.y * 32;
  const int lx = threadIdx.x, ly = threadIdx.y;
  const float* s = src + (size_t)b * C * L;
  #pragma unroll
  for (int i = 0; i < 32; i += 8)
    tile[ly + i][lx] = s[(size_t)(c0 + ly + i) * L + t0 + lx];
  __syncthreads();
  float* d = dst + (size_t)b * L * C;
  #pragma unroll
  for (int i = 0; i < 32; i += 8)
    d[(size_t)(t0 + ly + i) * C + c0 + lx] = tile[lx][ly + i];
}

// ---------------- fused main kernel ----------------
// grid = (L/32, B, K), block = 256 (8 waves).
// Phase 0: TDM-stage B panel  c[b, :, t0:t0+32] (bf16, padded rows) into LDS.
// Phase 1: c_proj [512 x 32] = W[k] @ panel via bf16 WMMA (8 WMMA / 8 A-loads / iter).
// Phase 2: pos/neg log-sigmoid scores from 11 gathered dots per t (bf16 c_proj in LDS).
__global__ __launch_bounds__(256) void k_cpc_step(
    const unsigned short* __restrict__ wbf,   // [K][C][C] bf16 (c contiguous)
    const unsigned short* __restrict__ ct,    // [B][L][C] bf16 (channel contiguous)
    const float* __restrict__ zt,             // [B][L][C] f32  (channel contiguous)
    const float* __restrict__ bias,           // [K][C] f32
    const int* __restrict__ num_neg_p,
    float* __restrict__ partials)             // [nblocks][2] (pos, neg)
{
  // overlaid: phase 0/1 = B panel (32 x 1040B), phase 1-store/2 = c_proj bf16 (32 x 1040B)
  __shared__ __align__(16) char s_mem[NT * LDSB_STRIDE];
  __shared__ float s_bias[CDIM];
  __shared__ float s_pos[8], s_neg[8];

  const int tid  = threadIdx.x;
  const int wave = tid >> 5;
  const int lane = tid & 31;
  const int lr   = lane & 15;
  const int lh   = lane >> 4;

  const int t0 = blockIdx.x * NT;
  const int b  = blockIdx.y;
  const int k  = blockIdx.z;
  const int T  = LDIM - k;
  const int bid = blockIdx.x + gridDim.x * (blockIdx.y + gridDim.y * blockIdx.z);

  if (t0 >= T) {                               // uniform per block
    if (tid == 0) { partials[2 * bid] = 0.f; partials[2 * bid + 1] = 0.f; }
    return;
  }

  s_bias[tid]       = bias[k * CDIM + tid];
  s_bias[tid + 256] = bias[k * CDIM + tid + 256];

  // ---- Phase 0: stage B panel into LDS
  const unsigned short* gsrc = ct + ((size_t)b * LDIM + t0) * CDIM;
#ifdef HAVE_TDM
  if (wave == 0) {
    const unsigned long long ga = (unsigned long long)gsrc;
    const unsigned ldsoff = (unsigned)(unsigned long long)(void*)s_mem;
    // D# group0: count=1, lds_addr, global_addr[56:0], type=2
    u32x4 g0 = { 1u, ldsoff, (unsigned)ga,
                 (unsigned)((ga >> 32) & 0x01FFFFFFu) | (2u << 30) };
    // D# group1: data_size=2B, pad_enable, pad_interval=256dw, pad_amount=4dw,
    //            tensor_dim0=512, tensor_dim1=32, tile_dim0=512, tile_dim1=32,
    //            tensor_dim0_stride=512
    i32x8 g1 = { 0x07D10000,      // mask=0 | ds=1<<16 | pad_en<<20 | 7<<22 | 3<<25
                 0x02000000,      // tensor_dim0[15:0]=512 << 16
                 0x00200000,      // tensor_dim1[15:0]=32 << 16
                 0x02000000,      // tile_dim0=512 << 16
                 32,              // tile_dim1=32
                 512,             // tensor_dim0_stride[31:0]=512
                 0, 0 };
    i32x4 gz4 = { 0, 0, 0, 0 };
    i32x8 gz8 = { 0, 0, 0, 0, 0, 0, 0, 0 };
    __builtin_amdgcn_tensor_load_to_lds(g0, g1, gz4, gz4, gz8, 0);
    __builtin_amdgcn_s_wait_tensorcnt((short)0);
  }
#else
  for (int idx = tid; idx < NT * (CDIM / 8); idx += 256) {
    const int row = idx >> 6, ch16 = idx & 63;         // 64 chunks of 8 bf16 per row
    uint4 v = *(const uint4*)(gsrc + (size_t)row * CDIM + ch16 * 8);
    *(uint4*)(s_mem + row * LDSB_STRIDE + ch16 * 16) = v;
  }
#endif
  __syncthreads();

  // ---- Phase 1: WMMA GEMM. Wave owns rows [wave*64, wave*64+64) x 32 cols.
  const unsigned short* Wk = wbf + (size_t)k * CDIM * CDIM;
  const int mbase = wave * 64;

  v8f acc[4][2];
  #pragma unroll
  for (int j = 0; j < 4; ++j)
    #pragma unroll
    for (int nt = 0; nt < 2; ++nt) acc[j][nt] = (v8f){};

  #pragma unroll 2
  for (int kk = 0; kk < CDIM; kk += 32) {
    // B operands from LDS: lane (lr,lh) col nt*16+lr, K chunk [kk+lh*16, +16)
    BfVec bm[2];
    #pragma unroll
    for (int nt = 0; nt < 2; ++nt) {
      const char* bp = s_mem + (nt * 16 + lr) * LDSB_STRIDE + (kk + lh * 16) * 2;
      bm[nt].q[0] = *(const uint4*)(bp);
      bm[nt].q[1] = *(const uint4*)(bp + 16);
    }
    #pragma unroll
    for (int j = 0; j < 4; ++j) {
      const unsigned short* ap =
          Wk + (size_t)(mbase + j * 16 + lr) * CDIM + kk + lh * 16;
      if (kk + 32 < CDIM) __builtin_prefetch(ap + 32, 0, 0);
      BfVec am;
      am.q[0] = *(const uint4*)(ap);
      am.q[1] = *(const uint4*)(ap + 8);
      #pragma unroll
      for (int nt = 0; nt < 2; ++nt)
        acc[j][nt] = __builtin_amdgcn_wmma_f32_16x16x32_bf16(
            false, am.v, false, bm[nt].v, (short)0, acc[j][nt], false, false);
    }
  }

  __syncthreads();   // everyone done reading the B panel; s_mem is reused below

  // Park c_proj (+bias) in LDS as bf16, [t][d] rows of stride 1040B.
  // C/D layout: vgpr v, lane -> (M = v + 8*lh, N = lr)
  #pragma unroll
  for (int j = 0; j < 4; ++j) {
    #pragma unroll
    for (int nt = 0; nt < 2; ++nt) {
      const int tl = nt * 16 + lr;
      #pragma unroll
      for (int v = 0; v < 8; v += 2) {
        const int d = mbase + j * 16 + lh * 8 + v;   // even
        const unsigned lo = f2bf(acc[j][nt][v]     + s_bias[d]);
        const unsigned hi = f2bf(acc[j][nt][v + 1] + s_bias[d + 1]);
        *(unsigned*)(s_mem + tl * LDSB_STRIDE + d * 2) = lo | (hi << 16);
      }
    }
  }
  __syncthreads();

  // ---- Phase 2: gathered score dots. Wave handles t-local {4*wave .. 4*wave+3}.
  int nn = *num_neg_p;
  if (nn > 16) nn = 16;
  float wpos = 0.f, wneg = 0.f;
  const float* zb = zt + (size_t)b * LDIM * CDIM;

  for (int ti = 0; ti < 4; ++ti) {
    const int tt = wave * 4 + ti;
    const int t  = t0 + tt;
    if (t >= T) continue;

    // hoist this lane's 16-channel c_proj column (bf16 -> f32 exact unpack)
    const unsigned* cw = (const unsigned*)(s_mem + tt * LDSB_STRIDE + lane * 32);
    float cpf[16];
    #pragma unroll
    for (int q = 0; q < 8; ++q) {
      const unsigned u = cw[q];
      cpf[2 * q]     = __uint_as_float(u << 16);
      cpf[2 * q + 1] = __uint_as_float(u & 0xFFFF0000u);
    }

    int slist[17];
    #pragma unroll
    for (int q = 0; q < 17; ++q) slist[q] = 0;
    slist[0] = t + k;  // positive

    // distinct random negatives in [0,T) — same distribution as top_k of iid uniforms
    unsigned seed = ((unsigned)k * gridDim.y + (unsigned)b) * (unsigned)LDIM + (unsigned)t;
    unsigned ctr = 0; int cnt = 0;
    while (cnt < nn && ctr < 1000u) {
      unsigned r = pcg(seed ^ (0x9E3779B9u * ++ctr));
      int cand = (int)(r % (unsigned)T);
      bool dup = false;
      for (int q = 0; q < cnt; ++q) dup |= (slist[1 + q] == cand);
      if (!dup) slist[1 + cnt++] = cand;
    }

    for (int si = 0; si <= nn; ++si) {
      const int s = slist[si];
      const float4* zr = (const float4*)(zb + (size_t)s * CDIM + lane * 16);
      float dacc = 0.f;
      #pragma unroll
      for (int q = 0; q < 4; ++q) {
        float4 zv = zr[q];
        dacc += zv.x * cpf[4 * q]     + zv.y * cpf[4 * q + 1] +
                zv.z * cpf[4 * q + 2] + zv.w * cpf[4 * q + 3];
      }
      #pragma unroll
      for (int off = 16; off; off >>= 1) dacc += __shfl_xor(dacc, off, 32);
      if (si == 0) wpos += log_sigmoid(dacc);
      else         wneg += log_sigmoid(-dacc);
    }
  }

  if (lane == 0) { s_pos[wave] = wpos; s_neg[wave] = wneg; }
  __syncthreads();
  if (tid == 0) {
    float p = 0.f, n = 0.f;
    #pragma unroll
    for (int w = 0; w < 8; ++w) { p += s_pos[w]; n += s_neg[w]; }
    partials[2 * bid]     = p;
    partials[2 * bid + 1] = n;
  }
}

// ---------------- deterministic final reduction ----------------
__global__ void k_finalize(const float* __restrict__ partials, int nblocks,
                           const int* __restrict__ num_neg_p,
                           float* __restrict__ out) {
  __shared__ float sp[256], sn[256];
  float p = 0.f, n = 0.f;
  for (int i = threadIdx.x; i < nblocks; i += 256) {
    p += partials[2 * i];
    n += partials[2 * i + 1];
  }
  sp[threadIdx.x] = p; sn[threadIdx.x] = n;
  __syncthreads();
  for (int s = 128; s; s >>= 1) {
    if (threadIdx.x < (unsigned)s) {
      sp[threadIdx.x] += sp[threadIdx.x + s];
      sn[threadIdx.x] += sn[threadIdx.x + s];
    }
    __syncthreads();
  }
  if (threadIdx.x == 0) {
    const float pos = sp[0], neg = sn[0];
    const float nnf = (float)(*num_neg_p);
    out[0] = -pos;
    out[1] = -neg;
    out[2] = -(pos + nnf * neg);
  }
}

// ---------------- launcher ----------------
extern "C" void kernel_launch(void* const* d_in, const int* in_sizes, int n_in,
                              void* d_out, int out_size, void* d_ws, size_t ws_size,
                              hipStream_t stream) {
  const float* z    = (const float*)d_in[0];   // [B,C,L]
  const float* c    = (const float*)d_in[1];   // [B,C,L]
  const float* W    = (const float*)d_in[2];   // [K,C,C]
  const float* bias = (const float*)d_in[3];   // [K,C]
  const int* num_neg_p = (const int*)d_in[5];

  const int C = in_sizes[2] / in_sizes[3];     // 512
  const int K = in_sizes[3] / C;               // 12
  const int L = LDIM;                          // 1024 (reference shape)
  const int B = in_sizes[0] / (C * L);         // 8

  // workspace layout: W bf16 | c^T bf16 | z^T f32 | partials
  char* ws = (char*)d_ws;
  size_t off = 0;
  unsigned short* wbf = (unsigned short*)(ws + off);
  off += (size_t)K * C * C * 2;  off = (off + 255) & ~(size_t)255;
  unsigned short* ct = (unsigned short*)(ws + off);
  off += (size_t)B * L * C * 2;  off = (off + 255) & ~(size_t)255;
  float* zt = (float*)(ws + off);
  off += (size_t)B * L * C * 4;  off = (off + 255) & ~(size_t)255;
  float* partials = (float*)(ws + off);

  const int nW = in_sizes[2];
  k_conv_w_bf16<<<(nW + 255) / 256, 256, 0, stream>>>(W, wbf, nW);

  dim3 tb(32, 8);
  k_transpose_c_bf16<<<dim3(L / 32, C / 32, B), tb, 0, stream>>>(c, ct, C, L);
  k_transpose_z_f32 <<<dim3(L / 32, C / 32, B), tb, 0, stream>>>(z, zt, C, L);

  dim3 grid(L / NT, B, K);
  k_cpc_step<<<grid, 256, 0, stream>>>(wbf, ct, zt, bias, num_neg_p, partials);

  const int nblocks = (L / NT) * B * K;
  k_finalize<<<1, 256, 0, stream>>>(partials, nblocks, num_neg_p, (float*)d_out);
}
